// StandardAttention_12120397709576
// MI455X (gfx1250) — matrix operand order repaired and checked by reading, not compiled, not run on hardware
//
#include <hip/hip_runtime.h>

// ---------------------------------------------------------------------------
// MHA forward for MI455X (gfx1250): bf16 WMMA (16x16x32) everywhere.
// x:[2,2048,1024] f32, w_qkv:[1024,3072], b_qkv:[3072], w_out:[1024,1024],
// b_out:[1024] -> out:[2,2048,1024] f32.
// ---------------------------------------------------------------------------

typedef __bf16 bf16;
typedef __attribute__((ext_vector_type(16))) __bf16 v16bf;
typedef __attribute__((ext_vector_type(8)))  __bf16 v8bf;
typedef __attribute__((ext_vector_type(4)))  __bf16 v4bf;
typedef __attribute__((ext_vector_type(8)))  float  v8f;

#define DEV __device__ __forceinline__

static constexpr int BATCH = 2;
static constexpr int NSEQ  = 2048;
static constexpr int DIM   = 1024;
static constexpr int DH    = 64;
static constexpr int MROWS = BATCH * NSEQ;   // 4096
static constexpr int D3    = 3 * DIM;        // 3072

DEV v8f wmma_bf16(v16bf a, v16bf b, v8f c) {
  // D = A(16x32) * B(32x16) + C, f32 accumulate
  return __builtin_amdgcn_wmma_f32_16x16x32_bf16(false, a, false, b,
                                                 (short)0, c, false, false);
}

// A-matrix 16x32 fragment from LDS (row-major, `stride` elems/row), base at
// (row0, kbase). ISA 7.12.2 A table: lane L holds row (L&15); lanes 0-15
// carry K {0..7, 16..23}, lanes 16-31 carry K {8..15, 24..31}; VGPR i holds
// the K pair (2i, 2i+1).
DEV v16bf fragA_ld(const bf16* base, int stride, int lane) {
  const bf16* p = base + (lane & 15) * stride + ((lane & 16) ? 8 : 0);
  v8bf lo = *(const v8bf*)p;          // K = kh .. kh+7
  v8bf hi = *(const v8bf*)(p + 16);   // K = kh+16 .. kh+23
  v16bf r;
#pragma unroll
  for (int i = 0; i < 8; ++i) { r[i] = lo[i]; r[8 + i] = hi[i]; }
  return r;
}

// B-matrix 32x16 fragment. LDS holds B^T row-major (row n = column n of B,
// contiguous in K). ISA B tables (16-bit, prefix of the 64x16 layout):
// lane L is column (L&15); lanes 0-15 carry K {0..7, 16..23}, lanes 16-31
// carry K {8..15, 24..31} (same K order as fragA_ld).
DEV v16bf fragB_ld(const bf16* base, int stride, int lane) {
  const bf16* p = base + (lane & 15) * stride + ((lane & 16) ? 8 : 0);
  v8bf lo = *(const v8bf*)p;          // K = kh .. kh+7
  v8bf hi = *(const v8bf*)(p + 16);   // K = kh+16 .. kh+23
  v16bf r;
#pragma unroll
  for (int i = 0; i < 8; ++i) { r[i] = lo[i]; r[8 + i] = hi[i]; }
  return r;
}

// ---------------------------------------------------------------------------
// Elementwise f32 -> bf16
// ---------------------------------------------------------------------------
__global__ void cvt_f32_bf16(const float* __restrict__ in,
                             bf16* __restrict__ out, int n) {
  int i = (blockIdx.x * blockDim.x + threadIdx.x) * 4;
  if (i + 3 < n) {
    float4 f = *(const float4*)(in + i);
    v4bf o; o[0] = (bf16)f.x; o[1] = (bf16)f.y; o[2] = (bf16)f.z; o[3] = (bf16)f.w;
    *(v4bf*)(out + i) = o;
  }
}

// ---------------------------------------------------------------------------
// Transpose + convert: in f32 [R][C] -> out bf16 [C][R]  (R,C multiples of 32)
// ---------------------------------------------------------------------------
__global__ void transpose_to_bf16(const float* __restrict__ in,
                                  bf16* __restrict__ out, int R, int C) {
  __shared__ float tile[32][33];
  const int r0 = blockIdx.y * 32, c0 = blockIdx.x * 32;
  const int tx = threadIdx.x, ty = threadIdx.y;   // 32 x 8
#pragma unroll
  for (int i = ty; i < 32; i += 8)
    tile[i][tx] = in[(size_t)(r0 + i) * C + c0 + tx];
  __syncthreads();
#pragma unroll
  for (int i = ty; i < 32; i += 8)
    out[(size_t)(c0 + i) * R + r0 + tx] = (bf16)tile[tx][i];
}

// ---------------------------------------------------------------------------
// bf16 WMMA GEMM: C[M][N] = A[M][K] * Bt[N][K]^T + bias.
// Block tile 128x128, K-step 32, 8 waves (2x4), each wave 64x32 (4x2 tiles).
// LDS stride 56 elems (112 B = 28 banks -> 16 distinct starting banks across
// the 16 fragment rows; rows stay 16-byte aligned for ds_load_b128).
// ---------------------------------------------------------------------------
template <bool OUT_F32>
__global__ __launch_bounds__(256)
void gemm_bf16(const bf16* __restrict__ A, const bf16* __restrict__ Bt,
               const float* __restrict__ bias, void* __restrict__ outp,
               int K, int lda, int ldb, int ldc) {
  constexpr int STR = 56;
  __shared__ bf16 As[128 * STR];
  __shared__ bf16 Bs[128 * STR];
  const int tid  = threadIdx.x;
  const int lane = tid & 31;
  const int wave = tid >> 5;
  const int wm = wave & 1, wn = wave >> 1;
  const int m0 = blockIdx.y * 128;
  const int n0 = blockIdx.x * 128;

  v8f acc[4][2];
#pragma unroll
  for (int tm = 0; tm < 4; ++tm)
#pragma unroll
    for (int tn = 0; tn < 2; ++tn)
#pragma unroll
      for (int v = 0; v < 8; ++v) acc[tm][tn][v] = 0.f;

  for (int k0 = 0; k0 < K; k0 += 32) {
    __syncthreads();
#pragma unroll
    for (int i = 0; i < 2; ++i) {              // 512 16B chunks / 256 threads
      int c = tid + i * 256;
      int r = c >> 2, cc = (c & 3) * 8;
      *(v8bf*)&As[r * STR + cc] = *(const v8bf*)&A[(size_t)(m0 + r) * lda + k0 + cc];
      *(v8bf*)&Bs[r * STR + cc] = *(const v8bf*)&Bt[(size_t)(n0 + r) * ldb + k0 + cc];
    }
    __syncthreads();

    v16bf af[4], bfr[2];
#pragma unroll
    for (int tm = 0; tm < 4; ++tm)
      af[tm] = fragA_ld(As + (wm * 64 + tm * 16) * STR, STR, lane);
#pragma unroll
    for (int tn = 0; tn < 2; ++tn)
      bfr[tn] = fragB_ld(Bs + (wn * 32 + tn * 16) * STR, STR, lane);
#pragma unroll
    for (int tm = 0; tm < 4; ++tm)
#pragma unroll
      for (int tn = 0; tn < 2; ++tn)
        acc[tm][tn] = wmma_bf16(af[tm], bfr[tn], acc[tm][tn]);
  }

  const int hi = lane >> 4, cl = lane & 15;
#pragma unroll
  for (int tm = 0; tm < 4; ++tm) {
#pragma unroll
    for (int tn = 0; tn < 2; ++tn) {
      const int gc = n0 + wn * 32 + tn * 16 + cl;
      const float bv = bias[gc];
#pragma unroll
      for (int v = 0; v < 8; ++v) {
        const int gr = m0 + wm * 64 + tm * 16 + v + hi * 8;
        const float val = acc[tm][tn][v] + bv;
        if (OUT_F32) ((float*)outp)[(size_t)gr * ldc + gc] = val;
        else         ((bf16*)outp)[(size_t)gr * ldc + gc] = (bf16)val;
      }
    }
  }
}

// ---------------------------------------------------------------------------
// Flash attention: one block per (b, h, 64-row q-block); 4 waves x 16 q-rows.
// qkv: bf16 [4096][3072] (Q|K|V), out: bf16 [4096][1024].
// ---------------------------------------------------------------------------
__global__ __launch_bounds__(128)
void attn_fwd(const bf16* __restrict__ qkv, bf16* __restrict__ outb) {
  constexpr int STR = 72;                     // 64 + 8 pad (144 B rows)
  __shared__ bf16 Ks[64 * STR];               // K rows (n x d)
  __shared__ bf16 Vt[64 * STR];               // V transposed (d x n)
  __shared__ bf16 Ps[64 * STR];               // Q staging, then per-wave P
  const int tid = threadIdx.x, lane = tid & 31, wave = tid >> 5;
  const int hi = lane >> 4, cl = lane & 15;
  const int qb = blockIdx.x & 31;
  const int h  = (blockIdx.x >> 5) & 15;
  const int b  = blockIdx.x >> 9;
  const size_t rowQ0 = (size_t)(b * NSEQ + qb * 64);
  const int qcol = h * DH, kcol = DIM + h * DH, vcol = 2 * DIM + h * DH;

  // Stage Q tile (64x64) and grab this wave's two A fragments.
  bf16* Qs = Ps;
#pragma unroll
  for (int i = 0; i < 4; ++i) {
    int c = tid + i * 128;
    int r = c >> 3, cc = (c & 7) * 8;
    *(v8bf*)&Qs[r * STR + cc] = *(const v8bf*)&qkv[(rowQ0 + r) * D3 + qcol + cc];
  }
  __syncthreads();
  v16bf qf[2];
  qf[0] = fragA_ld(Qs + (wave * 16) * STR,      STR, lane);
  qf[1] = fragA_ld(Qs + (wave * 16) * STR + 32, STR, lane);

  float m[8], l[8];
  v8f o[4];
#pragma unroll
  for (int v = 0; v < 8; ++v) { m[v] = -INFINITY; l[v] = 0.f; }
#pragma unroll
  for (int dt = 0; dt < 4; ++dt)
#pragma unroll
    for (int v = 0; v < 8; ++v) o[dt][v] = 0.f;

  for (int kb = 0; kb < NSEQ / 64; ++kb) {
    __syncthreads();
    const size_t rowK0 = (size_t)(b * NSEQ + kb * 64);
#pragma unroll
    for (int i = 0; i < 4; ++i) {
      int c = tid + i * 128;
      int r = c >> 3, cc = (c & 7) * 8;
      *(v8bf*)&Ks[r * STR + cc] = *(const v8bf*)&qkv[(rowK0 + r) * D3 + kcol + cc];
      v8bf vv = *(const v8bf*)&qkv[(rowK0 + r) * D3 + vcol + cc];
#pragma unroll
      for (int j = 0; j < 8; ++j) Vt[(cc + j) * STR + r] = vv[j];  // transpose
    }
    __syncthreads();

    // S = Q * K^T : B operand is K^T, i.e. B^T rows = K rows in Ks.
    v8f s[4];
#pragma unroll
    for (int nt = 0; nt < 4; ++nt) {
      v8f z = {0.f, 0.f, 0.f, 0.f, 0.f, 0.f, 0.f, 0.f};
      z = wmma_bf16(qf[0], fragB_ld(Ks + nt * 16 * STR,      STR, lane), z);
      z = wmma_bf16(qf[1], fragB_ld(Ks + nt * 16 * STR + 32, STR, lane), z);
      s[nt] = z;
    }
#pragma unroll
    for (int nt = 0; nt < 4; ++nt)
#pragma unroll
      for (int v = 0; v < 8; ++v) s[nt][v] *= 0.125f;   // DH^-0.5

    // Online softmax (row r = v + 8*hi lives in VGPR v; cols across 16 lanes).
    float mn[8];
#pragma unroll
    for (int v = 0; v < 8; ++v) {
      float x = fmaxf(fmaxf(s[0][v], s[1][v]), fmaxf(s[2][v], s[3][v]));
#pragma unroll
      for (int d = 1; d < 16; d <<= 1) x = fmaxf(x, __shfl_xor(x, d, 32));
      mn[v] = fmaxf(m[v], x);
    }
    bf16* Pw = Ps + wave * 16 * STR;
    float rs[8];
#pragma unroll
    for (int v = 0; v < 8; ++v) rs[v] = 0.f;
#pragma unroll
    for (int nt = 0; nt < 4; ++nt)
#pragma unroll
      for (int v = 0; v < 8; ++v) {
        float p = exp2f((s[nt][v] - mn[v]) * 1.44269504f);
        rs[v] += p;
        Pw[(v + 8 * hi) * STR + nt * 16 + cl] = (bf16)p;  // C-layout -> LDS
      }
#pragma unroll
    for (int v = 0; v < 8; ++v) {
#pragma unroll
      for (int d = 1; d < 16; d <<= 1) rs[v] += __shfl_xor(rs[v], d, 32);
      const float corr = exp2f((m[v] - mn[v]) * 1.44269504f);
      l[v] = l[v] * corr + rs[v];
      m[v] = mn[v];
#pragma unroll
      for (int dt = 0; dt < 4; ++dt) o[dt][v] *= corr;
    }

    // O += P * V.  P re-read from LDS in A-layout (same-wave DS ops are
    // in-order); V fragments are B operands read from the transposed Vt.
    v16bf pf0 = fragA_ld(Pw,      STR, lane);
    v16bf pf1 = fragA_ld(Pw + 32, STR, lane);
#pragma unroll
    for (int dt = 0; dt < 4; ++dt) {
      o[dt] = wmma_bf16(pf0, fragB_ld(Vt + dt * 16 * STR,      STR, lane), o[dt]);
      o[dt] = wmma_bf16(pf1, fragB_ld(Vt + dt * 16 * STR + 32, STR, lane), o[dt]);
    }
  }

  // Normalize and store bf16 into [4096][1024] for the output projection.
  const size_t orow0 = rowQ0 + wave * 16;
#pragma unroll
  for (int v = 0; v < 8; ++v) {
    const float inv = 1.0f / l[v];
    const size_t r = orow0 + v + 8 * hi;
#pragma unroll
    for (int dt = 0; dt < 4; ++dt)
      outb[r * DIM + h * DH + dt * 16 + cl] = (bf16)(o[dt][v] * inv);
  }
}

// ---------------------------------------------------------------------------
extern "C" void kernel_launch(void* const* d_in, const int* in_sizes, int n_in,
                              void* d_out, int out_size, void* d_ws, size_t ws_size,
                              hipStream_t stream) {
  (void)in_sizes; (void)n_in; (void)out_size; (void)ws_size;
  const float* x      = (const float*)d_in[0];
  const float* w_qkv  = (const float*)d_in[1];
  const float* b_qkv  = (const float*)d_in[2];
  const float* w_out  = (const float*)d_in[3];
  const float* b_out  = (const float*)d_in[4];
  float* out = (float*)d_out;

  char* ws = (char*)d_ws;
  size_t off = 0;
  auto alloc = [&](size_t bytes) {
    char* p = ws + off;
    off += (bytes + 255) & ~(size_t)255;
    return p;
  };
  bf16* xb    = (bf16*)alloc((size_t)MROWS * DIM * sizeof(bf16));   //  8 MB
  bf16* wqkvT = (bf16*)alloc((size_t)D3 * DIM * sizeof(bf16));      //  6 MB
  bf16* woutT = (bf16*)alloc((size_t)DIM * DIM * sizeof(bf16));     //  2 MB
  bf16* qkvb  = (bf16*)alloc((size_t)MROWS * D3 * sizeof(bf16));    // 24 MB
  bf16* attnb = (bf16*)alloc((size_t)MROWS * DIM * sizeof(bf16));   //  8 MB

  // 1. precision conversion / weight transposes
  cvt_f32_bf16<<<(MROWS * DIM) / (256 * 4), 256, 0, stream>>>(x, xb, MROWS * DIM);
  transpose_to_bf16<<<dim3(D3 / 32, DIM / 32), dim3(32, 8), 0, stream>>>(w_qkv, wqkvT, DIM, D3);
  transpose_to_bf16<<<dim3(DIM / 32, DIM / 32), dim3(32, 8), 0, stream>>>(w_out, woutT, DIM, DIM);

  // 2. QKV projection: [4096,1024] x [1024,3072] + bias -> bf16 qkv
  gemm_bf16<false><<<dim3(D3 / 128, MROWS / 128), 256, 0, stream>>>(
      xb, wqkvT, b_qkv, qkvb, DIM, DIM, DIM, D3);

  // 3. flash attention per (b, h, q-block)
  attn_fwd<<<BATCH * 16 * (NSEQ / 64), 128, 0, stream>>>(qkvb, attnb);

  // 4. output projection: [4096,1024] x [1024,1024] + bias -> f32 out
  gemm_bf16<true><<<dim3(DIM / 128, MROWS / 128), 256, 0, stream>>>(
      attnb, woutT, b_out, out, DIM, DIM, DIM, DIM);
}
